// GeneratorNetwork_49606872269425
// MI455X (gfx1250) — matrix operand order repaired and checked by reading, not compiled, not run on hardware
//
#include <hip/hip_runtime.h>
#include <hip/hip_bf16.h>

typedef __attribute__((ext_vector_type(16))) __bf16        v16bf;
typedef __attribute__((ext_vector_type(8)))  float         v8f;
typedef __attribute__((ext_vector_type(4)))  unsigned int  u32x4;

// guaranteed-global (addrspace 1) vector pointer for the weight streams
typedef const u32x4 __attribute__((address_space(1)))* gvec4p;

union BFrag { u32x4 u[2]; v16bf v; };
union AccU  { float f[8]; v8f v; };

__device__ __forceinline__ float sigmoidf_(float x) { return 1.0f / (1.0f + __expf(-x)); }

// ---------------------------------------------------------------------------
// Prep: convert Whh0 / Wih1 / Whh1 (each 1024x256 f32, row-major, gates=N dim)
// to bf16 and swizzle into WMMA 16x16x32 B-fragment layout:
//   fragment (kt,nt): 32 lanes x 8 dwords; lane = 16*hi + nl holds
//   n = nt*16+nl, k = kt*32 + (v>>2)*16 + hi*8 + 2*(v&3) (+1 in high half of the dword).
// ws layout: ws[mat*131072 + (kt*64+nt)*256 + lane*8 + v]
// ---------------------------------------------------------------------------
__global__ void prep_weights(const float* __restrict__ Whh0,
                             const float* __restrict__ Wih1,
                             const float* __restrict__ Whh1,
                             unsigned int* __restrict__ ws) {
  int id = blockIdx.x * blockDim.x + threadIdx.x;
  if (id >= 3 * 131072) return;
  int mat = id >> 17;
  int rem = id & 131071;
  int fragIdx = rem >> 8;          // kt*64 + nt
  int dw = rem & 255;              // lane*8 + v
  int lane = dw >> 3, v = dw & 7;
  int kt = fragIdx >> 6, nt = fragIdx & 63;
  int hi = lane >> 4, nl = lane & 15;
  int n = nt * 16 + nl;
  int k = kt * 32 + (v >> 2) * 16 + hi * 8 + (v & 3) * 2;
  const float* src = (mat == 0) ? Whh0 : (mat == 1) ? Wih1 : Whh1;
  float e0 = src[n * 256 + k];
  float e1 = src[n * 256 + k + 1];
  union { __bf16 b; unsigned short s; } c0, c1;
  c0.b = (__bf16)e0; c1.b = (__bf16)e1;
  ws[id] = ((unsigned int)c1.s << 16) | (unsigned int)c0.s;
}

// ---------------------------------------------------------------------------
// Main persistent recurrent kernel: 64 WGs x 16 batch rows, 8 waves each.
// Wave w owns h channels [32w, 32w+32) -> N-tiles {2w, 2w+1} and gate tiles
// nt = g*16 + 2w + t for gates g in {i,f,g,o}.
// ---------------------------------------------------------------------------
__global__ __launch_bounds__(256) void lstm_gen(
    const float* __restrict__ noise, const float* __restrict__ Wup,
    const float* __restrict__ bup,   const float* __restrict__ Wih0,
    const float* __restrict__ bih0,  const float* __restrict__ bhh0,
    const float* __restrict__ bih1,  const float* __restrict__ bhh1,
    const float* __restrict__ gamma_a, const float* __restrict__ beta_a,
    const float* __restrict__ Wa,      const float* __restrict__ ba,
    const float* __restrict__ gamma_w, const float* __restrict__ beta_w,
    const float* __restrict__ Ww,      const float* __restrict__ bw,
    const unsigned int* __restrict__ wsWhh0,
    const unsigned int* __restrict__ wsWih1,
    const unsigned int* __restrict__ wsWhh1,
    const int* __restrict__ nseg_p, float* __restrict__ out)
{
  __shared__ __align__(16) __bf16 hA0[2][16][256];   // layer-0 h, double buffered
  __shared__ __align__(16) __bf16 hA1[2][16][256];   // layer-1 h, double buffered
  __shared__ __align__(16) float  h1f[16][256];      // layer-1 h (f32) for GN/proj
  __shared__ float bias0s[1024], bias1s[1024];
  __shared__ float W0L[1024][2];
  __shared__ float gaL[256], beaL[256], waL[256], gwL[256], bewL[256], wwL[256];
  __shared__ float coordsL[16][2];
  __shared__ float gnMu[16][4], gnRs[16][4];

  const int tid  = threadIdx.x;
  const int w    = tid >> 5;
  const int lane = tid & 31;
  const int hi   = lane >> 4;
  const int ml   = lane & 15;
  const int r0   = blockIdx.x * 16;
  const int nseg = nseg_p[0];
  const float ba0 = ba[0], bw0 = bw[0];

  // ---- stage parameters in LDS
  for (int i = tid; i < 1024; i += 256) {
    bias0s[i] = bih0[i] + bhh0[i];
    bias1s[i] = bih1[i] + bhh1[i];
    W0L[i][0] = Wih0[i * 2 + 0];
    W0L[i][1] = Wih0[i * 2 + 1];
  }
  {
    int i = tid;
    gaL[i] = gamma_a[i]; beaL[i] = beta_a[i]; waL[i] = Wa[i];
    gwL[i] = gamma_w[i]; bewL[i] = beta_w[i]; wwL[i] = Ww[i];
  }
  if (tid < 32) coordsL[tid >> 1][tid & 1] = 0.0f;

  // ---- idea = elu(noise @ Wup^T + bup) for this WG's 16 rows
  {
    int row = tid >> 4, cbase = tid & 15;
    const float* nz = noise + (size_t)(r0 + row) * 128;
    for (int j = 0; j < 16; ++j) {
      int c = cbase + 16 * j;
      const float* wr = Wup + c * 128;
      float s = bup[c];
      for (int k = 0; k < 128; ++k) s += nz[k] * wr[k];
      float e = (s > 0.f) ? s : (__expf(s) - 1.f);
      h1f[row][c]      = e;             // stash for c0 init
      hA0[0][row][c]   = (__bf16)e;     // h0 init = idea
      hA1[0][row][c]   = (__bf16)0.0f;  // h1 init = 0
    }
  }
  __syncthreads();

  // ---- per-wave recurrent cell state: c0 = idea, c1 = 0
  float c0s[2][8], c1s[2][8];
  AccU  acc[2][4];
  #pragma unroll
  for (int tt = 0; tt < 2; ++tt) {
    int n = (2 * w + tt) * 16 + ml;
    #pragma unroll
    for (int r = 0; r < 8; ++r) {
      c0s[tt][r] = h1f[r + 8 * hi][n];
      c1s[tt][r] = 0.0f;
    }
  }
  __syncthreads();

  const unsigned long long laneByte = (unsigned long long)(lane * 8) * 4ull;

  for (int t = 0; t < nseg; ++t) {
    const int p = t & 1, q = p ^ 1;

    // Launder the weight base addresses once per step so LICM cannot hoist the
    // (step-invariant) fragment loads out of the recurrence loop and spill
    // the whole weight set to scratch. Keeping them as 64-bit integers and
    // dereferencing through addrspace(1) guarantees global_load codegen
    // (no FLAT, no DScnt coupling, no 64-bit vaddr pair).
    unsigned long long wh0a = (unsigned long long)wsWhh0;
    unsigned long long wi1a = (unsigned long long)wsWih1;
    unsigned long long wh1a = (unsigned long long)wsWhh1;
    asm volatile("" : "+s"(wh0a), "+s"(wi1a), "+s"(wh1a));
    wh0a += laneByte; wi1a += laneByte; wh1a += laneByte;

    // ================= layer 0: gates = coords@Wih0^T + h0@Whh0^T + b =======
    #pragma unroll
    for (int tt = 0; tt < 2; ++tt)
      #pragma unroll
      for (int g = 0; g < 4; ++g) {
        int n = (g * 16 + 2 * w + tt) * 16 + ml;
        float w0 = W0L[n][0], w1 = W0L[n][1], bb = bias0s[n];
        #pragma unroll
        for (int r = 0; r < 8; ++r) {
          int m = r + 8 * hi;
          acc[tt][g].f[r] = bb + coordsL[m][0] * w0 + coordsL[m][1] * w1;
        }
      }
    #pragma unroll
    for (int kt = 0; kt < 8; ++kt) {
      BFrag a;
      a.u[0] = *(const u32x4*)&hA0[p][ml][kt * 32 + hi * 8];
      a.u[1] = *(const u32x4*)&hA0[p][ml][kt * 32 + 16 + hi * 8];
      #pragma unroll
      for (int tt = 0; tt < 2; ++tt) {
        BFrag b[4];
        #pragma unroll
        for (int g = 0; g < 4; ++g) {
          int nt = g * 16 + 2 * w + tt;
          gvec4p bp = (gvec4p)(wh0a + (unsigned long long)((kt * 64 + nt) * 1024));
          b[g].u[0] = bp[0];
          b[g].u[1] = bp[1];
        }
        #pragma unroll
        for (int g = 0; g < 4; ++g)
          acc[tt][g].v = __builtin_amdgcn_wmma_f32_16x16x32_bf16(
              false, a.v, false, b[g].v, (short)0, acc[tt][g].v, false, false);
      }
    }
    #pragma unroll
    for (int tt = 0; tt < 2; ++tt) {
      int n = (2 * w + tt) * 16 + ml;
      #pragma unroll
      for (int r = 0; r < 8; ++r) {
        float iv = sigmoidf_(acc[tt][0].f[r]);
        float fv = sigmoidf_(acc[tt][1].f[r]);
        float gv = tanhf(acc[tt][2].f[r]);
        float ov = sigmoidf_(acc[tt][3].f[r]);
        float cn = fv * c0s[tt][r] + iv * gv;
        c0s[tt][r] = cn;
        hA0[q][r + 8 * hi][n] = (__bf16)(ov * tanhf(cn));
      }
    }
    __syncthreads();

    // ===== layer 1: gates = h0_cur@Wih1^T + h1_prev@Whh1^T + b ==============
    #pragma unroll
    for (int tt = 0; tt < 2; ++tt)
      #pragma unroll
      for (int g = 0; g < 4; ++g) {
        float bb = bias1s[(g * 16 + 2 * w + tt) * 16 + ml];
        #pragma unroll
        for (int r = 0; r < 8; ++r) acc[tt][g].f[r] = bb;
      }
    #pragma unroll
    for (int kt = 0; kt < 8; ++kt) {
      BFrag ax, ah;
      ax.u[0] = *(const u32x4*)&hA0[q][ml][kt * 32 + hi * 8];
      ax.u[1] = *(const u32x4*)&hA0[q][ml][kt * 32 + 16 + hi * 8];
      ah.u[0] = *(const u32x4*)&hA1[p][ml][kt * 32 + hi * 8];
      ah.u[1] = *(const u32x4*)&hA1[p][ml][kt * 32 + 16 + hi * 8];
      #pragma unroll
      for (int tt = 0; tt < 2; ++tt) {
        BFrag b[4];
        #pragma unroll
        for (int g = 0; g < 4; ++g) {
          int nt = g * 16 + 2 * w + tt;
          gvec4p bp = (gvec4p)(wi1a + (unsigned long long)((kt * 64 + nt) * 1024));
          b[g].u[0] = bp[0];
          b[g].u[1] = bp[1];
        }
        #pragma unroll
        for (int g = 0; g < 4; ++g)
          acc[tt][g].v = __builtin_amdgcn_wmma_f32_16x16x32_bf16(
              false, ax.v, false, b[g].v, (short)0, acc[tt][g].v, false, false);
        #pragma unroll
        for (int g = 0; g < 4; ++g) {
          int nt = g * 16 + 2 * w + tt;
          gvec4p bp = (gvec4p)(wh1a + (unsigned long long)((kt * 64 + nt) * 1024));
          b[g].u[0] = bp[0];
          b[g].u[1] = bp[1];
        }
        #pragma unroll
        for (int g = 0; g < 4; ++g)
          acc[tt][g].v = __builtin_amdgcn_wmma_f32_16x16x32_bf16(
              false, ah.v, false, b[g].v, (short)0, acc[tt][g].v, false, false);
      }
    }
    #pragma unroll
    for (int tt = 0; tt < 2; ++tt) {
      int n = (2 * w + tt) * 16 + ml;
      #pragma unroll
      for (int r = 0; r < 8; ++r) {
        float iv = sigmoidf_(acc[tt][0].f[r]);
        float fv = sigmoidf_(acc[tt][1].f[r]);
        float gv = tanhf(acc[tt][2].f[r]);
        float ov = sigmoidf_(acc[tt][3].f[r]);
        float cn = fv * c1s[tt][r] + iv * gv;
        c1s[tt][r] = cn;
        float hn = ov * tanhf(cn);
        int m = r + 8 * hi;
        hA1[q][m][n] = (__bf16)hn;
        h1f[m][n]    = hn;
      }
    }
    __syncthreads();

    // ===== GroupNorm stats: 16 rows x 4 groups of 64 channels ===============
    if (tid < 64) {
      int row = tid & 15, grp = tid >> 4;
      const float* hp = &h1f[row][grp * 64];
      float s = 0.f, ss = 0.f;
      for (int i = 0; i < 64; ++i) { float v = hp[i]; s += v; ss += v * v; }
      float mu = s * (1.0f / 64.0f);
      float var = ss * (1.0f / 64.0f) - mu * mu;
      gnMu[row][grp] = mu;
      gnRs[row][grp] = rsqrtf(var + 1e-5f);
    }
    __syncthreads();

    // ===== projections -> coords + output (wave w handles rows 2w, 2w+1) ===
    {
      int row = 2 * w + hi;
      float da = 0.f, dwd = 0.f;
      #pragma unroll
      for (int j = 0; j < 16; ++j) {
        int c = ml + 16 * j;
        int grp = c >> 6;
        float xn = (h1f[row][c] - gnMu[row][grp]) * gnRs[row][grp];
        da  += (xn * gaL[c] + beaL[c]) * waL[c];
        dwd += (xn * gwL[c] + bewL[c]) * wwL[c];
      }
      #pragma unroll
      for (int off = 8; off; off >>= 1) {   // xor<16 keeps rows separate
        da  += __shfl_xor(da,  off, 32);
        dwd += __shfl_xor(dwd, off, 32);
      }
      if (ml == 0) {
        float angle = tanhf(da + ba0);
        float width = sigmoidf_(dwd + bw0);
        coordsL[row][0] = angle;
        coordsL[row][1] = width;
        float* op = out + ((size_t)(r0 + row) * nseg + t) * 2;
        op[0] = angle; op[1] = width;
      }
    }
    __syncthreads();
  }
}

extern "C" void kernel_launch(void* const* d_in, const int* in_sizes, int n_in,
                              void* d_out, int out_size, void* d_ws, size_t ws_size,
                              hipStream_t stream) {
  const float* noise = (const float*)d_in[0];
  const float* Wup   = (const float*)d_in[1];
  const float* bup   = (const float*)d_in[2];
  const float* Wih0  = (const float*)d_in[3];
  const float* Whh0  = (const float*)d_in[4];
  const float* bih0  = (const float*)d_in[5];
  const float* bhh0  = (const float*)d_in[6];
  const float* Wih1  = (const float*)d_in[7];
  const float* Whh1  = (const float*)d_in[8];
  const float* bih1  = (const float*)d_in[9];
  const float* bhh1  = (const float*)d_in[10];
  const float* ga    = (const float*)d_in[11];
  const float* bea   = (const float*)d_in[12];
  const float* Wa    = (const float*)d_in[13];
  const float* ba    = (const float*)d_in[14];
  const float* gw    = (const float*)d_in[15];
  const float* bew   = (const float*)d_in[16];
  const float* Ww    = (const float*)d_in[17];
  const float* bw    = (const float*)d_in[18];
  const int*   nseg  = (const int*)d_in[19];

  unsigned int* ws = (unsigned int*)d_ws;   // needs 3 * 512 KB = 1.5 MB
  prep_weights<<<1536, 256, 0, stream>>>(Whh0, Wih1, Whh1, ws);
  lstm_gen<<<64, 256, 0, stream>>>(noise, Wup, bup, Wih0, bih0, bhh0,
                                   bih1, bhh1, ga, bea, Wa, ba, gw, bew, Ww, bw,
                                   ws, ws + 131072, ws + 262144,
                                   nseg, (float*)d_out);
}
